// GCNN_31774168056338
// MI455X (gfx1250) — matrix operand order repaired and checked by reading, not compiled, not run on hardware
//
#include <hip/hip_runtime.h>
#include <hip/hip_bf16.h>
#include <stdint.h>

// ---------------------------------------------------------------------------
// Types for CDNA5 WMMA (wave32): bf16 16x16x32, f32 accumulate
// ---------------------------------------------------------------------------
typedef __attribute__((ext_vector_type(16))) __bf16        v16bf;
typedef __attribute__((ext_vector_type(8)))  float         v8f;
typedef __attribute__((ext_vector_type(8)))  unsigned int  v8u;

#define SLOPE 0.2f

// fp32 -> bf16, round-to-nearest-even
static __device__ __forceinline__ unsigned short f2bf(float f) {
    unsigned int u = __float_as_uint(f);
    unsigned int r = u + 0x7FFFu + ((u >> 16) & 1u);
    return (unsigned short)(r >> 16);
}

// Order-preserving encode of float into u32 so u32-max == float max
static __device__ __forceinline__ unsigned int fenc(float f) {
    unsigned int u = __float_as_uint(f);
    return (u & 0x80000000u) ? ~u : (u | 0x80000000u);
}
static __device__ __forceinline__ float fdec(unsigned int u) {
    unsigned int v = (u & 0x80000000u) ? (u ^ 0x80000000u) : ~u;
    return __uint_as_float(v);
}
#define NEG_INF_ENC 0x007FFFFFu   // fenc(-inf)

static __device__ __forceinline__ float atomic_add_f32(float* p, float v) {
    return __hip_atomic_fetch_add(p, v, __ATOMIC_RELAXED, __HIP_MEMORY_SCOPE_AGENT);
}
static __device__ __forceinline__ unsigned int atomic_max_u32(unsigned int* p, unsigned int v) {
    return __hip_atomic_fetch_max(p, v, __ATOMIC_RELAXED, __HIP_MEMORY_SCOPE_AGENT);
}

// ---------------------------------------------------------------------------
// fp32 -> bf16 convert (optionally fused ReLU)
// ---------------------------------------------------------------------------
__global__ void k_cvt_bf16(const float* __restrict__ src, unsigned short* __restrict__ dst,
                           int n, int relu) {
    int i = blockIdx.x * blockDim.x + threadIdx.x;
    if (i >= n) return;
    float v = src[i];
    if (relu) v = v > 0.f ? v : 0.f;
    dst[i] = f2bf(v);
}

// Wt[n*D + k] = bf16(W[k*D + n])  (transpose so B-fragment loads are contiguous)
__global__ void k_prep_wt(const float* __restrict__ W, unsigned short* __restrict__ Wt, int D) {
    int i = blockIdx.x * blockDim.x + threadIdx.x;
    if (i >= D * D) return;
    int n = i / D, k = i % D;
    Wt[i] = f2bf(W[(size_t)k * D + n]);
}

// OUT[node, c] = b[c]   (bias folded into the aggregation accumulator)
__global__ void k_init_out(float* __restrict__ OUT, const float* __restrict__ b, int n, int D) {
    int i = blockIdx.x * blockDim.x + threadIdx.x;
    if (i >= n) return;
    OUT[i] = b[i % D];
}

__global__ void k_init_node(unsigned int* __restrict__ menc, float* __restrict__ den, int N) {
    int i = blockIdx.x * blockDim.x + threadIdx.x;
    if (i >= N) return;
    menc[i] = NEG_INF_ENC;
    den[i]  = 0.f;
}

// ---------------------------------------------------------------------------
// H[N,D] = A_bf16[N,D] @ W  via v_wmma_f32_16x16x32_bf16
// BT is W transposed: BT[n*D + k] = W[k][n]
//
// Block = 8 waves, one 64-column group (blockIdx.y). The 64 x DD bf16 weight
// panel (32 KB for DD=256) is staged in LDS once and shared by all 8 waves;
// B fragments come from ds_load_b128 (DScnt), A fragments from global
// (LOADcnt) -- independent counters, so WMMAs don't stall on global latency
// for B. Each wave computes 32 rows x 64 cols = 8 accumulators: per K-step
// 4 global b128 + 8 ds b128 feed 8 back-to-back WMMAs.
// ---------------------------------------------------------------------------
template <int DD>
__global__ void k_gemm_wmma_lds(const unsigned short* __restrict__ A,
                                const unsigned short* __restrict__ BT,
                                float* __restrict__ H, int Nrows) {
    __shared__ unsigned int Bsh[64 * (DD / 2)];     // 64 cols x DD K (bf16 pairs)

    const int tid  = threadIdx.x;
    const int lane = tid & 31;
    const int wave = tid >> 5;
    const int n0   = blockIdx.y << 6;

    // ---- cooperative copy: 64 contiguous BT rows -> LDS (contiguous 32KB) ----
    {
        const uint4* __restrict__ src4 = (const uint4*)(BT + (size_t)n0 * DD);
        uint4* dst4 = (uint4*)Bsh;
        constexpr int cnt = 64 * DD * 2 / 16;       // uint4 elements
        #pragma unroll
        for (int i = 0; i < cnt / 256; ++i)
            dst4[tid + i * 256] = src4[tid + i * 256];
    }
    __syncthreads();

    const int l15 = lane & 15;
    const bool hi = lane >= 16;
    const int m0  = (blockIdx.x * 8 + wave) << 5;   // 32 rows per wave

    // A row pointers for the two 16-row tiles (clamped; stores are guarded)
    int ar0 = m0 + l15;        if (ar0 >= Nrows) ar0 = Nrows - 1;
    int ar1 = m0 + 16 + l15;   if (ar1 >= Nrows) ar1 = Nrows - 1;
    const unsigned int* __restrict__ Ap0 =
        (const unsigned int*)(A + (size_t)ar0 * DD) + (hi ? 8 : 0);
    const unsigned int* __restrict__ Ap1 =
        (const unsigned int*)(A + (size_t)ar1 * DD) + (hi ? 8 : 0);

    // LDS B fragment bases (u32 units), one per 16-col tile
    const unsigned int* Bl0 = Bsh + (  0 + l15) * (DD / 2) + (hi ? 8 : 0);
    const unsigned int* Bl1 = Bsh + ( 16 + l15) * (DD / 2) + (hi ? 8 : 0);
    const unsigned int* Bl2 = Bsh + ( 32 + l15) * (DD / 2) + (hi ? 8 : 0);
    const unsigned int* Bl3 = Bsh + ( 48 + l15) * (DD / 2) + (hi ? 8 : 0);

    v8f c00 = {}, c01 = {}, c02 = {}, c03 = {};
    v8f c10 = {}, c11 = {}, c12 = {}, c13 = {};

    #pragma unroll
    for (int k0 = 0; k0 < DD; k0 += 32) {
        const int kk = k0 >> 1;                     // u32 offset for this K step

        v8u a0u, a1u, b0u, b1u, b2u, b3u;
        #pragma unroll
        for (int r = 0; r < 8; ++r) a0u[r] = Ap0[kk + r];
        #pragma unroll
        for (int r = 0; r < 8; ++r) a1u[r] = Ap1[kk + r];
        #pragma unroll
        for (int r = 0; r < 8; ++r) b0u[r] = Bl0[kk + r];
        #pragma unroll
        for (int r = 0; r < 8; ++r) b1u[r] = Bl1[kk + r];
        #pragma unroll
        for (int r = 0; r < 8; ++r) b2u[r] = Bl2[kk + r];
        #pragma unroll
        for (int r = 0; r < 8; ++r) b3u[r] = Bl3[kk + r];

        const v16bf a0 = __builtin_bit_cast(v16bf, a0u);
        const v16bf a1 = __builtin_bit_cast(v16bf, a1u);
        const v16bf b0 = __builtin_bit_cast(v16bf, b0u);
        const v16bf b1 = __builtin_bit_cast(v16bf, b1u);
        const v16bf b2 = __builtin_bit_cast(v16bf, b2u);
        const v16bf b3 = __builtin_bit_cast(v16bf, b3u);

        c00 = __builtin_amdgcn_wmma_f32_16x16x32_bf16(false, a0, false, b0, (short)0, c00, false, false);
        c01 = __builtin_amdgcn_wmma_f32_16x16x32_bf16(false, a0, false, b1, (short)0, c01, false, false);
        c02 = __builtin_amdgcn_wmma_f32_16x16x32_bf16(false, a0, false, b2, (short)0, c02, false, false);
        c03 = __builtin_amdgcn_wmma_f32_16x16x32_bf16(false, a0, false, b3, (short)0, c03, false, false);
        c10 = __builtin_amdgcn_wmma_f32_16x16x32_bf16(false, a1, false, b0, (short)0, c10, false, false);
        c11 = __builtin_amdgcn_wmma_f32_16x16x32_bf16(false, a1, false, b1, (short)0, c11, false, false);
        c12 = __builtin_amdgcn_wmma_f32_16x16x32_bf16(false, a1, false, b2, (short)0, c12, false, false);
        c13 = __builtin_amdgcn_wmma_f32_16x16x32_bf16(false, a1, false, b3, (short)0, c13, false, false);
    }

    // ---- store: tile VGPR r -> row (hi? 8+r : r), col = lane&15 ----
    #pragma unroll
    for (int mt = 0; mt < 2; ++mt) {
        const int mrow = m0 + mt * 16 + (hi ? 8 : 0);
        #pragma unroll
        for (int r = 0; r < 8; ++r) {
            const int m = mrow + r;
            if (m < Nrows) {
                float* Hp = H + (size_t)m * DD + n0 + l15;
                if (mt == 0) { Hp[0] = c00[r]; Hp[16] = c01[r]; Hp[32] = c02[r]; Hp[48] = c03[r]; }
                else         { Hp[0] = c10[r]; Hp[16] = c11[r]; Hp[32] = c12[r]; Hp[48] = c13[r]; }
            }
        }
    }
}

// Generic runtime-D fallback (register-only, rolled loops)
__global__ void k_gemm_wmma_gen(const unsigned short* __restrict__ A,
                                const unsigned short* __restrict__ BT,
                                float* __restrict__ H, int Nrows, int D) {
    const int lane = threadIdx.x & 31;
    const int wave = threadIdx.x >> 5;
    const int cgroups = D >> 6;
    const int mtiles  = (Nrows + 15) >> 4;
    const int ntasks  = mtiles * cgroups;
    const int task    = blockIdx.x * 8 + wave;
    if (task >= ntasks) return;

    const int m0 = (task / cgroups) << 4;
    const int n0 = (task % cgroups) << 6;
    const int l15 = lane & 15;
    const bool hi = lane >= 16;

    int arow = m0 + l15;
    if (arow >= Nrows) arow = Nrows - 1;
    const unsigned int* Ap = (const unsigned int*)(A + (size_t)arow * D) + (hi ? 8 : 0);

    v8f c0 = {}, c1 = {}, c2 = {}, c3 = {};
    for (int k0 = 0; k0 < D; k0 += 32) {
        const int kk = k0 >> 1;
        v8u au;
        #pragma unroll
        for (int r = 0; r < 8; ++r) au[r] = Ap[kk + r];
        const v16bf a = __builtin_bit_cast(v16bf, au);
        #pragma unroll
        for (int j = 0; j < 4; ++j) {
            const unsigned int* Bp =
                (const unsigned int*)(BT + (size_t)(n0 + j * 16 + l15) * D) + (hi ? 8 : 0);
            v8u bu;
            #pragma unroll
            for (int r = 0; r < 8; ++r) bu[r] = Bp[kk + r];
            const v16bf b = __builtin_bit_cast(v16bf, bu);
            if (j == 0) c0 = __builtin_amdgcn_wmma_f32_16x16x32_bf16(false, a, false, b, (short)0, c0, false, false);
            if (j == 1) c1 = __builtin_amdgcn_wmma_f32_16x16x32_bf16(false, a, false, b, (short)0, c1, false, false);
            if (j == 2) c2 = __builtin_amdgcn_wmma_f32_16x16x32_bf16(false, a, false, b, (short)0, c2, false, false);
            if (j == 3) c3 = __builtin_amdgcn_wmma_f32_16x16x32_bf16(false, a, false, b, (short)0, c3, false, false);
        }
    }

    const int mrow = m0 + (hi ? 8 : 0);
    #pragma unroll
    for (int r = 0; r < 8; ++r) {
        const int m = mrow + r;
        if (m < Nrows) {
            float* Hp = H + (size_t)m * D + n0 + l15;
            Hp[0]  = c0[r];
            Hp[16] = c1[r];
            Hp[32] = c2[r];
            Hp[48] = c3[r];
        }
    }
}

// ---------------------------------------------------------------------------
// es[i] = H[i,:]·a_src ; ed[i] = H[i,:]·a_dst   (one wave per node)
// ---------------------------------------------------------------------------
__global__ void k_dots(const float* __restrict__ H, const float* __restrict__ asrc,
                       const float* __restrict__ adst, float* __restrict__ es,
                       float* __restrict__ ed, int N, int D) {
    const int lane = threadIdx.x & 31;
    const int node = blockIdx.x * 8 + (threadIdx.x >> 5);
    if (node >= N) return;
    const float* h = H + (size_t)node * D;
    float s = 0.f, d = 0.f;
    for (int c = lane; c < D; c += 32) {
        float v = h[c];
        s += v * asrc[c];
        d += v * adst[c];
    }
    #pragma unroll
    for (int off = 16; off; off >>= 1) {
        s += __shfl_xor(s, off, 32);
        d += __shfl_xor(d, off, 32);
    }
    if (lane == 0) { es[node] = s; ed[node] = d; }
}

// edge pass 1: logit = leaky_relu(es[src]+ed[dst]); segment max into menc[dst]
__global__ void k_edge_max(const long long* __restrict__ ei, int E, int Etot,
                           const float* __restrict__ es, const float* __restrict__ ed,
                           float* __restrict__ elog, unsigned int* __restrict__ menc) {
    int e = blockIdx.x * blockDim.x + threadIdx.x;
    if (e >= Etot) return;
    int s, d;
    if (e < E) { s = (int)ei[e]; d = (int)ei[E + e]; } else { s = d = e - E; }
    float x = es[s] + ed[d];
    x = (x > 0.f) ? x : SLOPE * x;
    elog[e] = x;
    atomic_max_u32(&menc[d], fenc(x));
}

// edge pass 2: ex = exp(logit - m[dst]); den[dst] += ex (elog overwritten with ex)
__global__ void k_edge_exp(const long long* __restrict__ ei, int E, int Etot,
                           float* __restrict__ elog, const unsigned int* __restrict__ menc,
                           float* __restrict__ den) {
    int e = blockIdx.x * blockDim.x + threadIdx.x;
    if (e >= Etot) return;
    int d;
    if (e < E) { d = (int)ei[E + e]; } else { d = e - E; }
    float ex = __expf(elog[e] - fdec(menc[d]));
    elog[e] = ex;
    atomic_add_f32(&den[d], ex);
}

// edge pass 3: OUT[dst,:] += (ex/den[dst]) * H[src,:]   (one wave per edge)
// Lane handles 8 consecutive cols (2x float4 loads); wave covers one 1KB row.
__global__ void k_edge_aggr(const long long* __restrict__ ei, int E, int Etot,
                            const float* __restrict__ ex, const float* __restrict__ den,
                            const float* __restrict__ H, float* __restrict__ OUT, int D) {
    const int lane = threadIdx.x & 31;
    const int e = blockIdx.x * 8 + (threadIdx.x >> 5);
    if (e >= Etot) return;
    int s, d;
    if (e < E) { s = (int)ei[e]; d = (int)ei[E + e]; } else { s = d = e - E; }
    const float alpha = ex[e] / den[d];
    const float* hs = H + (size_t)s * D;
    float* od = OUT + (size_t)d * D;
    for (int c0 = lane * 8; c0 < D; c0 += 256) {
        const float4 v0 = *(const float4*)(hs + c0);
        const float4 v1 = *(const float4*)(hs + c0 + 4);
        atomic_add_f32(od + c0 + 0, alpha * v0.x);
        atomic_add_f32(od + c0 + 1, alpha * v0.y);
        atomic_add_f32(od + c0 + 2, alpha * v0.z);
        atomic_add_f32(od + c0 + 3, alpha * v0.w);
        atomic_add_f32(od + c0 + 4, alpha * v1.x);
        atomic_add_f32(od + c0 + 5, alpha * v1.y);
        atomic_add_f32(od + c0 + 6, alpha * v1.z);
        atomic_add_f32(od + c0 + 7, alpha * v1.w);
    }
}

// classifier: logits = relu(F) @ Wc + bc ; out = log_softmax (C == 2)
__global__ void k_classify(const float* __restrict__ F, const float* __restrict__ Wc,
                           const float* __restrict__ bc, float* __restrict__ out,
                           int N, int D) {
    const int lane = threadIdx.x & 31;
    const int node = blockIdx.x * 8 + (threadIdx.x >> 5);
    if (node >= N) return;
    const float* f = F + (size_t)node * D;
    float a0 = 0.f, a1 = 0.f;
    for (int c = lane; c < D; c += 32) {
        float v = f[c];
        v = v > 0.f ? v : 0.f;
        a0 += v * Wc[c * 2 + 0];
        a1 += v * Wc[c * 2 + 1];
    }
    #pragma unroll
    for (int off = 16; off; off >>= 1) {
        a0 += __shfl_xor(a0, off, 32);
        a1 += __shfl_xor(a1, off, 32);
    }
    if (lane == 0) {
        a0 += bc[0];
        a1 += bc[1];
        float m = fmaxf(a0, a1);
        float lse = m + __logf(__expf(a0 - m) + __expf(a1 - m));
        out[node * 2 + 0] = a0 - lse;
        out[node * 2 + 1] = a1 - lse;
    }
}

// ---------------------------------------------------------------------------
// Host-side orchestration
// ---------------------------------------------------------------------------
extern "C" void kernel_launch(void* const* d_in, const int* in_sizes, int n_in,
                              void* d_out, int out_size, void* d_ws, size_t ws_size,
                              hipStream_t stream) {
    const float*     x   = (const float*)d_in[0];
    const long long* ei  = (const long long*)d_in[1];   // int64 per reference
    const float*     W1  = (const float*)d_in[2];
    const float*     a1s = (const float*)d_in[3];
    const float*     a1d = (const float*)d_in[4];
    const float*     b1  = (const float*)d_in[5];
    const float*     W2  = (const float*)d_in[6];
    const float*     a2s = (const float*)d_in[7];
    const float*     a2d = (const float*)d_in[8];
    const float*     b2  = (const float*)d_in[9];
    const float*     Wc  = (const float*)d_in[10];
    const float*     bc  = (const float*)d_in[11];

    const int D    = in_sizes[3];        // 256
    const int N    = in_sizes[0] / D;    // 50000
    const int E    = in_sizes[1] / 2;    // 800000
    const int Etot = E + N;              // with self-loops

    // workspace carve-out (all buffers written before read)
    char* ws = (char*)d_ws;
    size_t off = 0;
    auto alloc = [&](size_t bytes) -> void* {
        off = (off + 255) & ~(size_t)255;
        void* p = ws + off;
        off += bytes;
        return p;
    };
    unsigned short* Xbf  = (unsigned short*)alloc((size_t)N * D * 2);
    float*          H    = (float*)alloc((size_t)N * D * 4);
    float*          OUT  = (float*)alloc((size_t)N * D * 4);
    unsigned short* Wt   = (unsigned short*)alloc((size_t)D * D * 2);
    float*          es   = (float*)alloc((size_t)N * 4);
    float*          ed   = (float*)alloc((size_t)N * 4);
    unsigned int*   menc = (unsigned int*)alloc((size_t)N * 4);
    float*          den  = (float*)alloc((size_t)N * 4);
    float*          elog = (float*)alloc((size_t)Etot * 4);

    const int ND = N * D;
    const dim3 B256(256);
    const int gND    = (ND + 255) / 256;
    const int gDD    = (D * D + 255) / 256;
    const int gN     = (N + 255) / 256;
    const int gNode8 = (N + 7) / 8;
    const int gE     = (Etot + 255) / 256;
    const int gE8    = (Etot + 7) / 8;

    auto gemm = [&](const unsigned short* Abf, const unsigned short* Bt, float* Hout) {
        if (D == 256) {
            const dim3 grid((N + 255) / 256, D / 64);   // 8 waves x 32 rows, 64 cols
            k_gemm_wmma_lds<256><<<grid, B256, 0, stream>>>(Abf, Bt, Hout, N);
        } else {
            const int mtiles = (N + 15) / 16;
            const int g = (mtiles * (D / 64) + 7) / 8;
            k_gemm_wmma_gen<<<g, B256, 0, stream>>>(Abf, Bt, Hout, N, D);
        }
    };

    // ---- layer 1 ----
    k_cvt_bf16 <<<gND,   B256, 0, stream>>>(x, Xbf, ND, 0);
    k_prep_wt  <<<gDD,   B256, 0, stream>>>(W1, Wt, D);
    k_init_out <<<gND,   B256, 0, stream>>>(OUT, b1, ND, D);
    k_init_node<<<gN,    B256, 0, stream>>>(menc, den, N);
    gemm(Xbf, Wt, H);
    k_dots     <<<gNode8,B256, 0, stream>>>(H, a1s, a1d, es, ed, N, D);
    k_edge_max <<<gE,    B256, 0, stream>>>(ei, E, Etot, es, ed, elog, menc);
    k_edge_exp <<<gE,    B256, 0, stream>>>(ei, E, Etot, elog, menc, den);
    k_edge_aggr<<<gE8,   B256, 0, stream>>>(ei, E, Etot, elog, den, H, OUT, D);

    // ---- layer 2 (ReLU fused into bf16 conversion of layer-1 output) ----
    k_cvt_bf16 <<<gND,   B256, 0, stream>>>(OUT, Xbf, ND, 1);
    k_prep_wt  <<<gDD,   B256, 0, stream>>>(W2, Wt, D);
    k_init_out <<<gND,   B256, 0, stream>>>(OUT, b2, ND, D);
    k_init_node<<<gN,    B256, 0, stream>>>(menc, den, N);
    gemm(Xbf, Wt, H);
    k_dots     <<<gNode8,B256, 0, stream>>>(H, a2s, a2d, es, ed, N, D);
    k_edge_max <<<gE,    B256, 0, stream>>>(ei, E, Etot, es, ed, elog, menc);
    k_edge_exp <<<gE,    B256, 0, stream>>>(ei, E, Etot, elog, menc, den);
    k_edge_aggr<<<gE8,   B256, 0, stream>>>(ei, E, Etot, elog, den, H, OUT, D);

    // ---- classifier + log_softmax (ReLU fused into the load) ----
    k_classify <<<gNode8,B256, 0, stream>>>(OUT, Wc, bc, (float*)d_out, N, D);
}
